// Muti_AFF_37580963840224
// MI455X (gfx1250) — matrix operand order repaired and checked by reading, not compiled, not run on hardware
//
#include <hip/hip_runtime.h>
#include <hip/hip_bf16.h>
#include <math.h>

typedef __attribute__((ext_vector_type(16))) _Float16 v16h;
typedef __attribute__((ext_vector_type(8)))  float    v8f;
typedef __attribute__((ext_vector_type(2)))  float    v2f;

#define DEV __device__ __forceinline__

DEV v8f wmma16(v16h a, v16h b, v8f c) {
  return __builtin_amdgcn_wmma_f32_16x16x32_f16(false, a, false, b, (short)0, c, false, false);
}
DEV v8f wmma4(v2f a, v2f b, v8f c) {
  return __builtin_amdgcn_wmma_f32_16x16x4_f32(false, a, false, b, (short)0, c, false, false);
}
DEV float sigm_(float x)  { return 1.0f / (1.0f + __expf(-x)); }
DEV float silu_(float x)  { return x * sigm_(x); }
DEV float softp_(float x) { return (x > 20.0f) ? x : logf(1.0f + __expf(x)); }

// ---------------- workspace layout (float offsets) ----------------
constexpr int OFF_GSUM = 0;                         // 8*64 per-(b,c) xa sums
constexpr int OFF_Y1P  = OFF_GSUM + 512;            // 64 parts * (16 sum + 16 sq)
constexpr int OFF_Y2P  = OFF_Y1P + 64 * 32;         // 64 parts * (64 sum + 64 sq)
constexpr int ZERO_CNT = OFF_Y2P + 64 * 128;        // region zeroed every launch
constexpr int OFF_BN1  = ZERO_CNT;                  // scale[16], shift[16]
constexpr int OFF_BN2  = OFF_BN1 + 32;              // scale[64], shift[64]
constexpr int OFF_XG   = OFF_BN2 + 128;             // [8][64]
constexpr int OFF_G    = OFF_XG + 512;              // [8][64] channel means
constexpr int OFF_ANEG = OFF_G + 512;               // 4*384*16  A = -exp(A_logs)
constexpr int OFF_MRPRE= OFF_ANEG + 4*384*16;       // [8][64][256] patch means
constexpr int OFF_Y1   = OFF_MRPRE + 8*64*256;      // [8][16][65536]
constexpr int OFF_XH   = OFF_Y1 + 8*16*65536;       // [8][256][192]
constexpr int OFF_XN   = OFF_XH + 8*256*192;        // [8][256][192]
constexpr int OFF_XC   = OFF_XN + 8*256*192;        // [8][384][256]
constexpr int OFF_ZB   = OFF_XC + 8*384*256;        // [8][256][384]
constexpr int OFF_XCS  = OFF_ZB + 8*256*384;        // [8][384][256]
constexpr int OFF_DEL  = OFF_XCS + 8*384*256;       // [8*4][256][384]
constexpr int OFF_BS   = OFF_DEL + 32*256*384;      // [8*4][256][16]
constexpr int OFF_CS   = OFF_BS + 32*256*16;        // [8*4][256][16]
constexpr int OFF_OY   = OFF_CS + 32*256*16;        // [8*4][256][384]
constexpr int OFF_YB   = OFF_OY + 32*256*384;       // [8][256][384]
constexpr int OFF_T    = OFF_YB + 8*256*384;        // [8][256][192]
constexpr int OFF_MRO  = OFF_T + 8*256*192;         // [8][64][256]

// ======================= pass 1: xa, means, conv1 (WMMA f16), y1 stats =======================
__global__ __launch_bounds__(256) void k1_stage(
    const float* __restrict__ x, const float* __restrict__ res,
    const float* __restrict__ w1, const float* __restrict__ b1,
    float* __restrict__ ws)
{
  __shared__ float xa[64 * 257];
  __shared__ float ssum[16], ssq[16];
  int t = threadIdx.x;
  int blk = blockIdx.x;
  int b = blk >> 8, patch = blk & 255;
  int ph = patch >> 4, pw = patch & 15;
  if (t < 16) { ssum[t] = 0.f; ssq[t] = 0.f; }
  for (int seg = t; seg < 1024; seg += 256) {
    int c = seg >> 4, r = seg & 15;
    int g = (b * 64 + c) * 65536 + (ph * 16 + r) * 256 + pw * 16;
    const float4* xp = (const float4*)(x + g);
    const float4* rp = (const float4*)(res + g);
    float* dst = xa + c * 257 + r * 16;
    #pragma unroll
    for (int i = 0; i < 4; ++i) {
      float4 a4 = xp[i], r4 = rp[i];
      dst[i*4+0] = a4.x + r4.x; dst[i*4+1] = a4.y + r4.y;
      dst[i*4+2] = a4.z + r4.z; dst[i*4+3] = a4.w + r4.w;
    }
  }
  __syncthreads();
  int w = t >> 5, l = t & 31, n = l & 15, hi = l >> 4;
  // patch means (for mr) + per-(b,c) totals (for global pool)
  for (int cc = 0; cc < 8; ++cc) {
    int c = w * 8 + cc;
    float s = 0.f;
    for (int i = l; i < 256; i += 32) s += xa[c * 257 + i];
    #pragma unroll
    for (int off = 16; off; off >>= 1) s += __shfl_xor(s, off);
    if (l == 0) {
      ws[OFF_MRPRE + (b * 64 + c) * 256 + patch] = s * (1.0f / 256.0f);
      atomicAdd(&ws[OFF_GSUM + b * 64 + c], s);
    }
  }
  // conv1: D(16 out x 16 px) = W1(16x64) * xa(64x16), K split in 2 steps of 32
  float sacc[8], qacc[8];
  #pragma unroll
  for (int v = 0; v < 8; ++v) { sacc[v] = 0.f; qacc[v] = 0.f; }
  for (int g2 = w * 2; g2 < w * 2 + 2; ++g2) {
    v8f c8 = {};
    int px = g2 * 16 + n;
    #pragma unroll
    for (int s = 0; s < 2; ++s) {
      v16h af, bf;
      #pragma unroll
      for (int j = 0; j < 8; ++j) {
        af[j]     = (_Float16)w1[n * 64 + s * 32 + 8 * hi + j];
        af[j + 8] = (_Float16)w1[n * 64 + s * 32 + 16 + 8 * hi + j];
      }
      #pragma unroll
      for (int j = 0; j < 16; ++j)
        bf[j] = (_Float16)xa[(s * 32 + (j < 8 ? 8 * hi + j : 8 + 8 * hi + j)) * 257 + px];
      c8 = wmma16(af, bf, c8);
    }
    int r = px >> 4, col = px & 15;
    #pragma unroll
    for (int v = 0; v < 8; ++v) {
      int o = v + 8 * hi;
      float y = c8[v] + b1[o];
      ws[OFF_Y1 + (b * 16 + o) * 65536 + (ph * 16 + r) * 256 + pw * 16 + col] = y;
      sacc[v] += y; qacc[v] += y * y;
    }
  }
  #pragma unroll
  for (int v = 0; v < 8; ++v) {
    #pragma unroll
    for (int off = 8; off; off >>= 1) {
      sacc[v] += __shfl_xor(sacc[v], off);
      qacc[v] += __shfl_xor(qacc[v], off);
    }
  }
  if (n == 0) {
    #pragma unroll
    for (int v = 0; v < 8; ++v) {
      atomicAdd(&ssum[v + 8 * hi], sacc[v]);
      atomicAdd(&ssq[v + 8 * hi], qacc[v]);
    }
  }
  __syncthreads();
  if (t < 16) {
    int part = blk & 63;
    atomicAdd(&ws[OFF_Y1P + part * 32 + t], ssum[t]);
    atomicAdd(&ws[OFF_Y1P + part * 32 + 16 + t], ssq[t]);
  }
}

// ======================= tiny: BN1 params, g means, global branch, A = -exp =======================
__global__ __launch_bounds__(256) void k2_setup(
    const float* lg1, const float* lbe1,
    const float* gw1, const float* gb1, const float* gg1, const float* gbe1,
    const float* gw2, const float* gb2, const float* gg2, const float* gbe2,
    const float* alogs, float* ws)
{
  __shared__ float yg1[128], zg1[128], yg2[512];
  int t = threadIdx.x;
  if (t < 16) {
    float s = 0.f, q = 0.f;
    for (int p = 0; p < 64; ++p) { s += ws[OFF_Y1P + p*32 + t]; q += ws[OFF_Y1P + p*32 + 16 + t]; }
    float inv = 1.0f / 524288.0f;
    float m = s * inv, var = q * inv - m * m;
    float sc = lg1[t] * rsqrtf(var + 1e-5f);
    ws[OFF_BN1 + t] = sc; ws[OFF_BN1 + 16 + t] = lbe1[t] - m * sc;
  }
  for (int i = t; i < 512; i += 256) ws[OFF_G + i] = ws[OFF_GSUM + i] * (1.0f / 65536.0f);
  for (int i = t; i < 4*384*16; i += 256) ws[OFF_ANEG + i] = -__expf(alogs[i]);
  __syncthreads();
  if (t < 128) {
    int b = t >> 4, ic = t & 15;
    float acc = gb1[ic];
    for (int c = 0; c < 64; ++c) acc += gw1[ic*64 + c] * ws[OFF_G + b*64 + c];
    yg1[t] = acc;
  }
  __syncthreads();
  if (t < 16) {
    float s = 0.f, q = 0.f;
    for (int b = 0; b < 8; ++b) { float v = yg1[b*16 + t]; s += v; q += v*v; }
    float m = s / 8.f, var = q / 8.f - m*m, r = rsqrtf(var + 1e-5f);
    for (int b = 0; b < 8; ++b) {
      float v = (yg1[b*16 + t] - m) * r * gg1[t] + gbe1[t];
      zg1[b*16 + t] = fmaxf(v, 0.f);
    }
  }
  __syncthreads();
  for (int i = t; i < 512; i += 256) {
    int b = i >> 6, c = i & 63;
    float acc = gb2[c];
    for (int ic = 0; ic < 16; ++ic) acc += gw2[c*16 + ic] * zg1[b*16 + ic];
    yg2[i] = acc;
  }
  __syncthreads();
  if (t < 64) {
    float s = 0.f, q = 0.f;
    for (int b = 0; b < 8; ++b) { float v = yg2[b*64 + t]; s += v; q += v*v; }
    float m = s / 8.f, var = q / 8.f - m*m, r = rsqrtf(var + 1e-5f);
    for (int b = 0; b < 8; ++b)
      ws[OFF_XG + b*64 + t] = (yg2[b*64 + t] - m) * r * gg2[t] + gbe2[t];
  }
}

// ======================= mr branch =======================
__global__ void k3_convin(const float* __restrict__ inw, float* __restrict__ ws) {
  int tid = blockIdx.x * 256 + threadIdx.x;
  if (tid >= 8*192*256) return;
  int p = tid & 255, rem = tid >> 8;
  int d = rem % 192, b = rem / 192;
  float acc = 0.f;
  for (int c = 0; c < 64; ++c) acc += inw[d*64 + c] * ws[OFF_MRPRE + (b*64 + c)*256 + p];
  ws[OFF_XH + (b*256 + p)*192 + d] = acc;
}

__global__ void k3_ln(const float* __restrict__ lng, const float* __restrict__ lnb,
                      float* __restrict__ ws) {
  int tid = blockIdx.x * 256 + threadIdx.x; if (tid >= 2048) return;
  const float* src = ws + OFF_XH + tid * 192;
  float s = 0.f, q = 0.f;
  for (int d = 0; d < 192; ++d) { float v = src[d]; s += v; q += v*v; }
  float m = s / 192.f, var = q / 192.f - m*m, r = rsqrtf(var + 1e-5f);
  float* dst = ws + OFF_XN + tid * 192;
  for (int d = 0; d < 192; ++d) dst[d] = (src[d] - m) * r * lng[d] + lnb[d];
}

// xz = xn @ ss_in_w.T  (192 -> 768) via WMMA f16; xc channel-major, z pixel-major
__global__ __launch_bounds__(256) void k3_ssin(const float* __restrict__ ssw, float* __restrict__ ws) {
  int b = blockIdx.x;
  int t = threadIdx.x, w = t >> 5, l = t & 31, n = l & 15, hi = l >> 4;
  const float* xn = ws + OFF_XN + b * 256 * 192;
  for (int tid = w; tid < 768; tid += 8) {
    int mt = tid >> 4, nt = tid & 15;
    int px = nt * 16 + n;
    int m = mt * 16 + n;
    v8f c8 = {};
    #pragma unroll
    for (int s = 0; s < 6; ++s) {
      v16h af, bf;
      #pragma unroll
      for (int j = 0; j < 8; ++j) {
        af[j]     = (_Float16)ssw[m*192 + s*32 + 8*hi + j];
        af[j + 8] = (_Float16)ssw[m*192 + s*32 + 16 + 8*hi + j];
      }
      #pragma unroll
      for (int j = 0; j < 16; ++j)
        bf[j] = (_Float16)xn[px*192 + s*32 + (j < 8 ? 8*hi + j : 8 + 8*hi + j)];
      c8 = wmma16(af, bf, c8);
    }
    #pragma unroll
    for (int v = 0; v < 8; ++v) {
      int f = mt*16 + v + 8*hi;
      float val = c8[v];
      if (f < 384) ws[OFF_XC + (b*384 + f)*256 + px] = val;
      else         ws[OFF_ZB + (b*256 + px)*384 + (f - 384)] = val;
    }
  }
}

__global__ void k3_dw(const float* __restrict__ cw, const float* __restrict__ cb,
                      float* __restrict__ ws) {
  int tid = blockIdx.x * 256 + threadIdx.x; if (tid >= 8*384*256) return;
  int p = tid & 255; int rem = tid >> 8; int c = rem % 384; int b = rem / 384;
  int h = p >> 4, wq = p & 15;
  const float* src = ws + OFF_XC + (b*384 + c)*256;
  float acc = cb[c];
  #pragma unroll
  for (int dh = -1; dh <= 1; ++dh)
    #pragma unroll
    for (int dw = -1; dw <= 1; ++dw) {
      int hh = h + dh, ww = wq + dw;
      if (hh >= 0 && hh < 16 && ww >= 0 && ww < 16)
        acc += cw[c*9 + (dh+1)*3 + (dw+1)] * src[hh*16 + ww];
    }
  ws[OFF_XCS + (b*384 + c)*256 + p] = silu_(acc);
}

// x_proj (384 -> 44) + dt projection + softplus, per direction k
__global__ __launch_bounds__(256) void k3_proj(const float* __restrict__ xpw,
                                               const float* __restrict__ dtw,
                                               const float* __restrict__ dtb,
                                               float* __restrict__ ws) {
  int bk = blockIdx.x; int b = bk >> 2, k = bk & 3; int l = threadIdx.x;
  int lt = (l & 15) * 16 + (l >> 4);
  int rl = 255 - l;
  int psrc = (k == 0) ? l : (k == 1) ? lt : (k == 2) ? rl : ((rl & 15)*16 + (rl >> 4));
  float acc[44];
  #pragma unroll
  for (int c = 0; c < 44; ++c) acc[c] = 0.f;
  const float* xs = ws + OFF_XCS + b * 384 * 256;
  for (int d = 0; d < 384; ++d) {
    float xv = xs[d * 256 + psrc];
    #pragma unroll
    for (int c = 0; c < 44; ++c) acc[c] += xpw[(k*44 + c)*384 + d] * xv;
  }
  #pragma unroll
  for (int i = 0; i < 16; ++i) {
    ws[OFF_BS + (bk*256 + l)*16 + i] = acc[12 + i];
    ws[OFF_CS + (bk*256 + l)*16 + i] = acc[28 + i];
  }
  for (int d = 0; d < 384; ++d) {
    float s = dtb[k*384 + d];
    #pragma unroll
    for (int r = 0; r < 12; ++r) s += dtw[(k*384 + d)*12 + r] * acc[r];
    ws[OFF_DEL + (bk*256 + l)*384 + d] = softp_(s);
  }
}

// selective scan: one block per (b,k); thread per channel d, 16-state in regs
__global__ __launch_bounds__(384) void k3_scan(const float* __restrict__ Ds, float* __restrict__ ws) {
  __shared__ float Bl[4096], Cl[4096];
  int bk = blockIdx.x; int b = bk >> 2, k = bk & 3; int d = threadIdx.x;
  for (int i = d; i < 4096; i += 384) { Bl[i] = ws[OFF_BS + bk*4096 + i]; Cl[i] = ws[OFF_CS + bk*4096 + i]; }
  __syncthreads();
  float A[16], h[16];
  #pragma unroll
  for (int n2 = 0; n2 < 16; ++n2) { A[n2] = ws[OFF_ANEG + (k*384 + d)*16 + n2]; h[n2] = 0.f; }
  float Dv = Ds[k*384 + d];
  const float* xs = ws + OFF_XCS + (b*384 + d)*256;
  for (int l = 0; l < 256; ++l) {
    int lt = (l & 15)*16 + (l >> 4);
    int rl = 255 - l;
    int psrc = (k == 0) ? l : (k == 1) ? lt : (k == 2) ? rl : ((rl & 15)*16 + (rl >> 4));
    float dt = ws[OFF_DEL + (bk*256 + l)*384 + d];
    float u = xs[psrc];
    float du = dt * u, y = 0.f;
    #pragma unroll
    for (int n2 = 0; n2 < 16; ++n2) {
      h[n2] = __expf(dt * A[n2]) * h[n2] + du * Bl[l*16 + n2];
      y += h[n2] * Cl[l*16 + n2];
    }
    ws[OFF_OY + (bk*256 + l)*384 + d] = y + Dv * u;
  }
}

// combine 4 directions, LN over DI, gate with silu(z)
__global__ void k3_comb(const float* __restrict__ ong, const float* __restrict__ onb,
                        float* __restrict__ ws) {
  int tid = blockIdx.x * 256 + threadIdx.x; if (tid >= 2048) return;
  int b = tid >> 8, l = tid & 255;
  int lt = (l & 15)*16 + (l >> 4);
  int o0 = ((b*4 + 0)*256 + l) * 384;
  int o1 = ((b*4 + 1)*256 + lt) * 384;
  int o2 = ((b*4 + 2)*256 + (255 - l)) * 384;
  int o3 = ((b*4 + 3)*256 + (255 - lt)) * 384;
  float* yb = ws + OFF_YB + (b*256 + l)*384;
  const float* oy = ws + OFF_OY;
  float s = 0.f, q = 0.f;
  for (int d = 0; d < 384; ++d) {
    float v = oy[o0 + d] + oy[o1 + d] + oy[o2 + d] + oy[o3 + d];
    yb[d] = v; s += v; q += v*v;
  }
  float m = s / 384.f, var = q / 384.f - m*m, r = rsqrtf(var + 1e-5f);
  const float* z = ws + OFF_ZB + (b*256 + l)*384;
  for (int d = 0; d < 384; ++d)
    yb[d] = ((yb[d] - m) * r * ong[d] + onb[d]) * silu_(z[d]);
}

// t = yb @ ss_out_w.T + xh  (384 -> 192) via WMMA
__global__ __launch_bounds__(256) void k3_ssout(const float* __restrict__ sow, float* __restrict__ ws) {
  int b = blockIdx.x;
  int t = threadIdx.x, w = t >> 5, l = t & 31, n = l & 15, hi = l >> 4;
  const float* yb = ws + OFF_YB + b * 256 * 384;
  for (int tid = w; tid < 192; tid += 8) {
    int mt = tid >> 4, nt = tid & 15;
    int px = nt * 16 + n, m = mt * 16 + n;
    v8f c8 = {};
    #pragma unroll
    for (int s = 0; s < 12; ++s) {
      v16h af, bf;
      #pragma unroll
      for (int j = 0; j < 8; ++j) {
        af[j]     = (_Float16)sow[m*384 + s*32 + 8*hi + j];
        af[j + 8] = (_Float16)sow[m*384 + s*32 + 16 + 8*hi + j];
      }
      #pragma unroll
      for (int j = 0; j < 16; ++j)
        bf[j] = (_Float16)yb[px*384 + s*32 + (j < 8 ? 8*hi + j : 8 + 8*hi + j)];
      c8 = wmma16(af, bf, c8);
    }
    #pragma unroll
    for (int v = 0; v < 8; ++v) {
      int d2 = mt*16 + v + 8*hi;
      ws[OFF_T + (b*256 + px)*192 + d2] = c8[v] + ws[OFF_XH + (b*256 + px)*192 + d2];
    }
  }
}

// mr_out = t @ out_w.T  (192 -> 64) via WMMA
__global__ __launch_bounds__(256) void k3_outw(const float* __restrict__ ow, float* __restrict__ ws) {
  int b = blockIdx.x;
  int t = threadIdx.x, w = t >> 5, l = t & 31, n = l & 15, hi = l >> 4;
  const float* tb = ws + OFF_T + b * 256 * 192;
  for (int tid = w; tid < 64; tid += 8) {
    int mt = tid >> 4, nt = tid & 15;
    int px = nt * 16 + n, m = mt * 16 + n;
    v8f c8 = {};
    #pragma unroll
    for (int s = 0; s < 6; ++s) {
      v16h af, bf;
      #pragma unroll
      for (int j = 0; j < 8; ++j) {
        af[j]     = (_Float16)ow[m*192 + s*32 + 8*hi + j];
        af[j + 8] = (_Float16)ow[m*192 + s*32 + 16 + 8*hi + j];
      }
      #pragma unroll
      for (int j = 0; j < 16; ++j)
        bf[j] = (_Float16)tb[px*192 + s*32 + (j < 8 ? 8*hi + j : 8 + 8*hi + j)];
      c8 = wmma16(af, bf, c8);
    }
    #pragma unroll
    for (int v = 0; v < 8; ++v) {
      int c = mt*16 + v + 8*hi;
      ws[OFF_MRO + (b*64 + c)*256 + px] = c8[v];
    }
  }
}

// ======================= pass 2: y2 = conv2(relu(bn1(y1))), global stats (WMMA f32 K=4) =======================
__global__ __launch_bounds__(256) void k5_y2stats(const float* __restrict__ w2,
                                                  const float* __restrict__ b2,
                                                  float* __restrict__ ws) {
  __shared__ float z[16 * 257];
  __shared__ float ssum[64], ssq[64];
  int t = threadIdx.x, blk = blockIdx.x;
  int b = blk >> 8, patch = blk & 255, ph = patch >> 4, pw = patch & 15;
  if (t < 64) { ssum[t] = 0.f; ssq[t] = 0.f; }
  {
    int o = t >> 4, r = t & 15;
    const float* src = ws + OFF_Y1 + (b*16 + o)*65536 + (ph*16 + r)*256 + pw*16;
    float sc = ws[OFF_BN1 + o], sh = ws[OFF_BN1 + 16 + o];
    #pragma unroll
    for (int i = 0; i < 16; ++i)
      z[o*257 + r*16 + i] = fmaxf(src[i]*sc + sh, 0.f);
  }
  __syncthreads();
  int w = t >> 5, l = t & 31, n = l & 15, hi = l >> 4;
  for (int tid = w; tid < 64; tid += 8) {
    int mt = tid >> 4, ng = tid & 15;
    int px = ng*16 + n;
    v8f c8 = {};
    #pragma unroll
    for (int s = 0; s < 4; ++s) {
      int k0 = s*4 + 2*hi;
      v2f a, bb;
      a[0] = w2[(mt*16 + n)*16 + k0]; a[1] = w2[(mt*16 + n)*16 + k0 + 1];
      bb[0] = z[k0*257 + px];         bb[1] = z[(k0 + 1)*257 + px];
      c8 = wmma4(a, bb, c8);
    }
    float sa[8], qa[8];
    #pragma unroll
    for (int v = 0; v < 8; ++v) {
      float y = c8[v] + b2[mt*16 + v + 8*hi];
      sa[v] = y; qa[v] = y*y;
    }
    #pragma unroll
    for (int v = 0; v < 8; ++v) {
      #pragma unroll
      for (int off = 8; off; off >>= 1) {
        sa[v] += __shfl_xor(sa[v], off);
        qa[v] += __shfl_xor(qa[v], off);
      }
    }
    if (n == 0) {
      #pragma unroll
      for (int v = 0; v < 8; ++v) {
        atomicAdd(&ssum[mt*16 + v + 8*hi], sa[v]);
        atomicAdd(&ssq[mt*16 + v + 8*hi], qa[v]);
      }
    }
  }
  __syncthreads();
  if (t < 64) {
    int part = blk & 63;
    atomicAdd(&ws[OFF_Y2P + part*128 + t], ssum[t]);
    atomicAdd(&ws[OFF_Y2P + part*128 + 64 + t], ssq[t]);
  }
}

__global__ void k6_bn2(const float* __restrict__ lg2, const float* __restrict__ lbe2,
                       float* __restrict__ ws) {
  int t = threadIdx.x; if (t >= 64) return;
  float s = 0.f, q = 0.f;
  for (int p = 0; p < 64; ++p) { s += ws[OFF_Y2P + p*128 + t]; q += ws[OFF_Y2P + p*128 + 64 + t]; }
  float inv = 1.0f / 524288.0f, m = s * inv, var = q * inv - m*m;
  float sc = lg2[t] * rsqrtf(var + 1e-5f);
  ws[OFF_BN2 + t] = sc; ws[OFF_BN2 + 64 + t] = lbe2[t] - m * sc;
}

// ======================= pass 3: final fuse =======================
__global__ __launch_bounds__(256) void k7_final(
    const float* __restrict__ x, const float* __restrict__ res,
    const float* __restrict__ w2, const float* __restrict__ b2,
    const float* __restrict__ ws, float* __restrict__ out)
{
  __shared__ float z[16 * 257];
  int t = threadIdx.x, blk = blockIdx.x;
  int b = blk >> 8, patch = blk & 255, ph = patch >> 4, pw = patch & 15;
  {
    int o = t >> 4, r = t & 15;
    const float* src = ws + OFF_Y1 + (b*16 + o)*65536 + (ph*16 + r)*256 + pw*16;
    float sc = ws[OFF_BN1 + o], sh = ws[OFF_BN1 + 16 + o];
    #pragma unroll
    for (int i = 0; i < 16; ++i)
      z[o*257 + r*16 + i] = fmaxf(src[i]*sc + sh, 0.f);
  }
  __syncthreads();
  int w = t >> 5, l = t & 31, n = l & 15, hi = l >> 4;
  for (int tid = w; tid < 64; tid += 8) {
    int mt = tid >> 4, ng = tid & 15;
    int px = ng*16 + n;
    v8f c8 = {};
    #pragma unroll
    for (int s = 0; s < 4; ++s) {
      int k0 = s*4 + 2*hi;
      v2f a, bb;
      a[0] = w2[(mt*16 + n)*16 + k0]; a[1] = w2[(mt*16 + n)*16 + k0 + 1];
      bb[0] = z[k0*257 + px];         bb[1] = z[(k0 + 1)*257 + px];
      c8 = wmma4(a, bb, c8);
    }
    int r = px >> 4, col = px & 15;
    #pragma unroll
    for (int v = 0; v < 8; ++v) {
      int o = mt*16 + v + 8*hi;
      float y = c8[v] + b2[o];
      float xl = y * ws[OFF_BN2 + o] + ws[OFF_BN2 + 64 + o];
      float wei = sigm_(xl + ws[OFF_XG + b*64 + o] + ws[OFF_MRO + (b*64 + o)*256 + patch]);
      int g = (b*64 + o)*65536 + (ph*16 + r)*256 + pw*16 + col;
      out[g] = 2.0f * (x[g] * wei + res[g] * (1.0f - wei));
    }
  }
}

extern "C" void kernel_launch(void* const* d_in, const int* in_sizes, int n_in,
                              void* d_out, int out_size, void* d_ws, size_t ws_size,
                              hipStream_t stream)
{
  (void)in_sizes; (void)n_in; (void)out_size; (void)ws_size;
  const float* x       = (const float*)d_in[0];
  const float* res     = (const float*)d_in[1];
  const float* l_w1    = (const float*)d_in[2];
  const float* l_b1    = (const float*)d_in[3];
  const float* l_g1    = (const float*)d_in[4];
  const float* l_be1   = (const float*)d_in[5];
  const float* l_w2    = (const float*)d_in[6];
  const float* l_b2    = (const float*)d_in[7];
  const float* l_g2    = (const float*)d_in[8];
  const float* l_be2   = (const float*)d_in[9];
  const float* g_w1    = (const float*)d_in[10];
  const float* g_b1    = (const float*)d_in[11];
  const float* g_g1    = (const float*)d_in[12];
  const float* g_be1   = (const float*)d_in[13];
  const float* g_w2    = (const float*)d_in[14];
  const float* g_b2    = (const float*)d_in[15];
  const float* g_g2    = (const float*)d_in[16];
  const float* g_be2   = (const float*)d_in[17];
  const float* in_w    = (const float*)d_in[18];
  const float* out_w   = (const float*)d_in[19];
  const float* ln_g    = (const float*)d_in[20];
  const float* ln_b    = (const float*)d_in[21];
  const float* ss_in_w = (const float*)d_in[22];
  const float* conv_w  = (const float*)d_in[23];
  const float* conv_b  = (const float*)d_in[24];
  const float* x_proj_w= (const float*)d_in[25];
  const float* dt_w    = (const float*)d_in[26];
  const float* dt_b    = (const float*)d_in[27];
  const float* A_logs  = (const float*)d_in[28];
  const float* Ds      = (const float*)d_in[29];
  const float* onorm_g = (const float*)d_in[30];
  const float* onorm_b = (const float*)d_in[31];
  const float* ss_out_w= (const float*)d_in[32];
  float* ws = (float*)d_ws;

  hipMemsetAsync(d_ws, 0, (size_t)ZERO_CNT * sizeof(float), stream);
  k1_stage<<<2048, 256, 0, stream>>>(x, res, l_w1, l_b1, ws);
  k2_setup<<<1, 256, 0, stream>>>(l_g1, l_be1, g_w1, g_b1, g_g1, g_be1,
                                  g_w2, g_b2, g_g2, g_be2, A_logs, ws);
  k3_convin<<<(8*192*256)/256, 256, 0, stream>>>(in_w, ws);
  k3_ln<<<8, 256, 0, stream>>>(ln_g, ln_b, ws);
  k3_ssin<<<8, 256, 0, stream>>>(ss_in_w, ws);
  k3_dw<<<(8*384*256)/256, 256, 0, stream>>>(conv_w, conv_b, ws);
  k3_proj<<<32, 256, 0, stream>>>(x_proj_w, dt_w, dt_b, ws);
  k3_scan<<<32, 384, 0, stream>>>(Ds, ws);
  k3_comb<<<8, 256, 0, stream>>>(onorm_g, onorm_b, ws);
  k3_ssout<<<8, 256, 0, stream>>>(ss_out_w, ws);
  k3_outw<<<8, 256, 0, stream>>>(out_w, ws);
  k5_y2stats<<<2048, 256, 0, stream>>>(l_w2, l_b2, ws);
  k6_bn2<<<1, 64, 0, stream>>>(l_g2, l_be2, ws);
  k7_final<<<2048, 256, 0, stream>>>(x, res, l_w2, l_b2, ws, (float*)d_out);
}
